// Node_only_random_agg_75900662055227
// MI455X (gfx1250) — matrix operand order repaired and checked by reading, not compiled, not run on hardware
//
#include <hip/hip_runtime.h>
#include <hip/hip_bf16.h>

// ---------------------------------------------------------------------------
// Types for CDNA5 WMMA (wave32): bf16 A/B (16 elems/lane), f32 accum (8/lane)
// ---------------------------------------------------------------------------
typedef __attribute__((ext_vector_type(16))) __bf16 v16bf;
typedef __attribute__((ext_vector_type(8)))  float  v8f;

// ---------------- bf16 <-> f32 helpers (bit-exact RNE, stored as u16) ------
__device__ __forceinline__ unsigned short f2bf(float f) {
  unsigned u = __float_as_uint(f);
  u += 0x7FFFu + ((u >> 16) & 1u);   // round-to-nearest-even
  return (unsigned short)(u >> 16);
}
__device__ __forceinline__ float bf2f(unsigned short h) {
  return __uint_as_float(((unsigned)h) << 16);
}
__device__ __forceinline__ void atomic_fadd(float* p, float v) {
  __hip_atomic_fetch_add(p, v, __ATOMIC_RELAXED, __HIP_MEMORY_SCOPE_AGENT);
}

// ---------------------------------------------------------------------------
// Small utility kernels
// ---------------------------------------------------------------------------
__global__ void fill_f32(float* __restrict__ p, float v, size_t n) {
  size_t i = (size_t)blockIdx.x * blockDim.x + threadIdx.x;
  size_t stride = (size_t)gridDim.x * blockDim.x;
  for (; i < n; i += stride) p[i] = v;
}

__global__ void deg_kernel(const int* __restrict__ dst, float* __restrict__ deg, int E) {
  int e = blockIdx.x * blockDim.x + threadIdx.x;
  if (e < E) atomic_fadd(&deg[dst[e]], 1.0f);
}

__global__ void dinv_kernel(float* __restrict__ deg, int n) {
  int i = blockIdx.x * blockDim.x + threadIdx.x;
  if (i < n) { float d = deg[i]; deg[i] = (d > 0.f) ? rsqrtf(d) : 0.f; }
}

__global__ void norm_kernel(const int* __restrict__ src, const int* __restrict__ dst,
                            const float* __restrict__ dinv, float* __restrict__ norm, int E) {
  int e = blockIdx.x * blockDim.x + threadIdx.x;
  if (e < E) norm[e] = dinv[src[e]] * dinv[dst[e]];
}

__global__ void f32_to_bf16(const float* __restrict__ in, unsigned short* __restrict__ out,
                            size_t n) {
  size_t i = (size_t)blockIdx.x * blockDim.x + threadIdx.x;
  size_t stride = (size_t)gridDim.x * blockDim.x;
  for (; i < n; i += stride) out[i] = f2bf(in[i]);
}

// W [din, dout] f32 row-major -> Wt [dout, din] bf16 (so B K-pairs are contiguous)
__global__ void w_transpose_bf16(const float* __restrict__ W, unsigned short* __restrict__ Wt,
                                 int din, int dout) {
  size_t total = (size_t)din * dout;
  size_t i = (size_t)blockIdx.x * blockDim.x + threadIdx.x;
  size_t stride = (size_t)gridDim.x * blockDim.x;
  for (; i < total; i += stride) {
    int r = (int)(i / dout);   // k
    int c = (int)(i % dout);   // n
    Wt[(size_t)c * din + r] = f2bf(W[i]);
  }
}

// ---------------------------------------------------------------------------
// bf16 WMMA GEMM:  C[M,Nout](bf16) = A[M,K](bf16) * Bt[Nout,K](bf16)^T
//
// Block = 8 waves stacked over M (128 rows) sharing ONE 16-column B tile.
// B slab (16 cols x 128 K = 4KB) staged in LDS with gfx1250 async copies
// (global_load_async_to_lds_b128 + s_wait_asynccnt), giving 8x B reuse.
// A rows stream from global as two contiguous b128 loads per lane.
// D layout follows CDNA5 ISA 7.12.2 (wave32); A/B use contiguous k = 16*half + s in both.
// Requires K % 128 == 0 (true here: 128/4096/2048/1024/512/256).
// ---------------------------------------------------------------------------
#define GEMM_KS 128

__global__ __launch_bounds__(256)
void gemm_bf16_wmma(const unsigned short* __restrict__ A,
                    const unsigned short* __restrict__ Bt,
                    unsigned short* __restrict__ C,
                    int M, int K, int Nout) {
  __shared__ __align__(32) unsigned short lds_b[16 * GEMM_KS];  // 4KB

  const int tid  = threadIdx.x;
  const int lane = tid & 31;
  const int wave = tid >> 5;
  const int row0 = blockIdx.y * 128 + wave * 16;
  const int col0 = blockIdx.x * 16;
  const int hl   = lane >> 4;       // half of the wave
  const int l16  = lane & 15;

  const bool rowvalid = (row0 < M);         // wave-uniform
  const int  m = rowvalid ? (row0 + l16) : 0;
  const int  n = col0 + l16;
  const bool nvalid = (n < Nout);

  const unsigned short* arow = A + (size_t)m * K;

  // --- async-staging assignment: thread -> (column nl, 16B chunk kc) ---
  const int nl = tid >> 4;                  // 0..15 local column
  const int kc = (tid & 15) << 3;           // 0..120 step 8 (bf16 elems = 16B)
  int bn = col0 + nl; if (bn >= Nout) bn = Nout - 1;   // clamp (cols masked at store)
  const unsigned short* bsrc_base = Bt + (size_t)bn * K + kc;
  const unsigned lds_dst = (unsigned)(uintptr_t)&lds_b[nl * GEMM_KS + kc];

  v8f acc = {};
  for (int k0 = 0; k0 < K; k0 += GEMM_KS) {
    // ---- stage B slab into LDS (256 threads x 16B = 4KB) ----
    {
      const unsigned short* g = bsrc_base + k0;
      asm volatile("global_load_async_to_lds_b128 %0, %1, off"
                   :: "v"(lds_dst), "v"(g) : "memory");
      asm volatile("s_wait_asynccnt 0" ::: "memory");
    }
    __syncthreads();

    if (rowvalid) {
      __builtin_prefetch(arow + k0 + GEMM_KS, 0, 3);
#pragma unroll
      for (int kk = 0; kk < GEMM_KS; kk += 32) {
        union { v16bf v; unsigned u[8]; uint4 q[2]; } a, b;
        // A 16x32 bf16 layout: lanes 0-15 K=0..15, lanes 16-31 K=16..31
        // (matches B) -> two contiguous 16B chunks.
        a.q[0] = *(const uint4*)(arow + k0 + kk + hl * 16);
        a.q[1] = *(const uint4*)(arow + k0 + kk + hl * 16 + 8);
        // B 32x16 bf16 layout: lane covers col l16; lanes 0-15 K=0..15,
        // lanes 16-31 K=16..31 -> one contiguous 32B LDS read.
        const unsigned short* lb = &lds_b[l16 * GEMM_KS + kk + hl * 16];
        b.q[0] = *(const uint4*)(lb);
        b.q[1] = *(const uint4*)(lb + 8);
        acc = __builtin_amdgcn_wmma_f32_16x16x32_bf16(
            /*neg_a=*/false, a.v, /*neg_b=*/false, b.v,
            /*c_mod=*/(short)0, acc, /*reuse_a=*/false, /*reuse_b=*/false);
      }
    }
    __syncthreads();
  }

  // D layout: VGPR r -> M=row0+r (lanes 0-15) / M=row0+r+8 (lanes 16-31), N=l16.
  if (rowvalid && nvalid) {
#pragma unroll
    for (int r = 0; r < 8; ++r) {
      int mm = row0 + r + hl * 8;
      C[(size_t)mm * Nout + n] = f2bf(acc[r]);
    }
  }
}

// ---------------------------------------------------------------------------
// Edge scatter: agg[dst] += bf2f(lin[src]) * norm   (block per edge)
// ---------------------------------------------------------------------------
__global__ void edge_scatter(const int* __restrict__ src, const int* __restrict__ dst,
                             const float* __restrict__ norm,
                             const unsigned short* __restrict__ lin,
                             float* __restrict__ agg, int dout, int E) {
  int e = blockIdx.x;
  if (e >= E) return;
  int s = src[e], d = dst[e];
  float w = norm[e];
  const unsigned short* ls = lin + (size_t)s * dout;
  float* ad = agg + (size_t)d * dout;
  for (int f = threadIdx.x; f < dout; f += blockDim.x)
    atomic_fadd(&ad[f], bf2f(ls[f]) * w);
}

// finalize: h = relu(agg + lin_self * dinv^2 + bias) -> bf16 (ping-pong)
__global__ void finalize_kernel(const float* __restrict__ agg,
                                const unsigned short* __restrict__ lin,
                                const float* __restrict__ dinv,
                                const float* __restrict__ bias,
                                unsigned short* __restrict__ hout,
                                int dout, size_t total) {
  size_t i = (size_t)blockIdx.x * blockDim.x + threadIdx.x;
  size_t stride = (size_t)gridDim.x * blockDim.x;
  for (; i < total; i += stride) {
    int node = (int)(i / dout);
    int f = (int)(i % dout);
    float di = dinv[node];
    float v = agg[i] + bf2f(lin[i]) * di * di + bias[f];
    hout[i] = f2bf(v > 0.f ? v : 0.f);
  }
}

// Final: mean over 50 gathered rows of h[:,200], then L2 normalize.
__global__ void path_mean_norm(const unsigned short* __restrict__ h,
                               const int* __restrict__ pidx,
                               float* __restrict__ out) {
  __shared__ float red[256];
  __shared__ float s_scale;
  int d = threadIdx.x;
  float mean = 0.f;
  if (d < 200) {
    float acc = 0.f;
    for (int j = 0; j < 50; ++j)
      acc += bf2f(h[(size_t)pidx[j] * 200 + d]);
    mean = acc * (1.0f / 50.0f);
  }
  red[threadIdx.x] = (d < 200) ? mean * mean : 0.f;
  __syncthreads();
  for (int s = 128; s > 0; s >>= 1) {
    if (threadIdx.x < s) red[threadIdx.x] += red[threadIdx.x + s];
    __syncthreads();
  }
  if (threadIdx.x == 0) s_scale = (red[0] > 0.f) ? rsqrtf(red[0]) : 0.f;
  __syncthreads();
  if (d < 200) out[d] = mean * s_scale;
}

// ---------------------------------------------------------------------------
// Host launcher
// ---------------------------------------------------------------------------
extern "C" void kernel_launch(void* const* d_in, const int* in_sizes, int n_in,
                              void* d_out, int out_size, void* d_ws, size_t ws_size,
                              hipStream_t stream) {
  (void)n_in; (void)out_size; (void)ws_size;
  const int N = 20000;
  const int E = in_sizes[1] / 2;
  static const int DIMS[7] = {128, 4096, 2048, 1024, 512, 256, 200};

  const float* x    = (const float*)d_in[0];
  const int*   ei   = (const int*)d_in[1];
  const int*   srcI = ei;
  const int*   dstI = ei + E;
  const int*   pidx = (const int*)d_in[2];

  // --- carve workspace ---
  char* ws = (char*)d_ws;
  size_t off = 0;
  auto carve = [&](size_t bytes) -> char* {
    char* p = ws + off;
    off = (off + bytes + 255) & ~(size_t)255;
    return p;
  };
  float* dinv = (float*)carve((size_t)N * sizeof(float));     // deg -> dinv in place
  float* norm = (float*)carve((size_t)E * sizeof(float));
  unsigned short* Wt[6];
  for (int l = 0; l < 6; ++l)
    Wt[l] = (unsigned short*)carve((size_t)DIMS[l] * DIMS[l + 1] * sizeof(unsigned short));
  unsigned short* hA  = (unsigned short*)carve((size_t)N * 4096 * sizeof(unsigned short));
  unsigned short* lin = (unsigned short*)carve((size_t)N * 4096 * sizeof(unsigned short));
  float*          agg = (float*)carve((size_t)N * 4096 * sizeof(float));

  const int T = 256;

  // --- GCN norm ---
  fill_f32<<<dim3(256), T, 0, stream>>>(dinv, 1.0f, (size_t)N);         // self-loop
  deg_kernel<<<dim3((E + T - 1) / T), T, 0, stream>>>(dstI, dinv, E);
  dinv_kernel<<<dim3((N + T - 1) / T), T, 0, stream>>>(dinv, N);
  norm_kernel<<<dim3((E + T - 1) / T), T, 0, stream>>>(srcI, dstI, dinv, norm, E);

  // --- one-time conversions ---
  f32_to_bf16<<<dim3(2048), T, 0, stream>>>(x, hA, (size_t)N * DIMS[0]);
  for (int l = 0; l < 6; ++l)
    w_transpose_bf16<<<dim3(2048), T, 0, stream>>>((const float*)d_in[3 + 2 * l],
                                                   Wt[l], DIMS[l], DIMS[l + 1]);

  // --- 6 GCN layers ---
  for (int l = 0; l < 6; ++l) {
    const int din = DIMS[l], dout = DIMS[l + 1];
    const float* bias = (const float*)d_in[4 + 2 * l];
    size_t total = (size_t)N * dout;

    dim3 ggrid((dout + 15) / 16, (N + 127) / 128);
    gemm_bf16_wmma<<<ggrid, 256, 0, stream>>>(hA, Wt[l], lin, N, din, dout);

    fill_f32<<<dim3(4096), T, 0, stream>>>(agg, 0.f, total);
    edge_scatter<<<dim3(E), T, 0, stream>>>(srcI, dstI, norm, lin, agg, dout, E);
    finalize_kernel<<<dim3(4096), T, 0, stream>>>(agg, lin, dinv, bias, hA, dout, total);
  }

  // --- path gather + mean + L2 normalize ---
  path_mean_norm<<<1, 256, 0, stream>>>(hA, pidx, (float*)d_out);
}
